// RecencyBiasAttention_81896436400623
// MI455X (gfx1250) — compile-verified, matrix operand order checked by reading, not run on hardware
//
#include <hip/hip_runtime.h>
#include <hip/hip_bf16.h>
#include <stdint.h>

typedef __attribute__((ext_vector_type(16))) _Float16 v16h;
typedef __attribute__((ext_vector_type(8)))  float    v8f;
typedef __attribute__((ext_vector_type(4)))  unsigned int v4u;
typedef __attribute__((ext_vector_type(8)))  int      v8i;
typedef __attribute__((ext_vector_type(4)))  int      v4i;

#define D_MODEL   1024
#define N_HEADS   16
#define HEAD_DIM  64
#define SEQ       2048
#define BATCH     4
#define NQKV      3072
#define LDS_STRIDE 34   // 32 + 2 halves pad -> conflict-free 16-row fragment reads

// Load a 16x32 f16 A/B fragment (ISA 7.12.2 layout) from a row-major [row][k]
// buffer. p must point at element [row_for_this_lane][half*8]. Per-VGPR K pairs:
// r<4 -> k = 2r (+half*8), r>=4 -> k = 2r+8 (+half*8).
__device__ __forceinline__ v16h load_ab_frag(const _Float16* p) {
  union { v16h v; uint32_t u[8]; } f;
#pragma unroll
  for (int r = 0; r < 8; ++r) {
    const int kof = 2 * r + ((r >= 4) ? 8 : 0);
    f.u[r] = *reinterpret_cast<const uint32_t*>(p + kof);
  }
  return f.v;
}

// Issue one TDM 2D tile load global->LDS (D# per ISA cdna5 ch8).
// data_size=1 (2 bytes/elem). Optionally pad LDS dest by 1 DWORD per 16 DWORDs
// so a 32-half tile row lands with a 34-half stride (bank-conflict-free).
__device__ __forceinline__ void tdm_load_2d(const _Float16* gptr, uint32_t lds_off,
                                            uint32_t tensor_d0, uint32_t tensor_d1,
                                            uint32_t tile_d0, uint32_t tile_d1,
                                            uint32_t row_stride, int pad) {
  const uint64_t ga = (uint64_t)(uintptr_t)gptr;
  v4u g0;
  g0[0] = 1u;                                   // count=1, user descriptor
  g0[1] = lds_off;                              // lds_addr (bytes)
  g0[2] = (uint32_t)ga;                         // global_addr[31:0]
  g0[3] = (uint32_t)((ga >> 32) & 0x01FFFFFFu)  // global_addr[56:32]
        | (2u << 30);                           // type = 2 ("image")
  v8i g1;
  uint32_t dw0 = (1u << 16);                    // data_size = 1 -> 2 bytes
  if (pad) dw0 |= (1u << 20)                    // pad_enable
              |  (3u << 22)                     // pad_interval: 16 DWORDs
              |  (0u << 25);                    // pad_amount:   1 DWORD
  g1[0] = (int)dw0;                             // workgroup_mask=0 (no cluster)
  g1[1] = (int)((tensor_d0 & 0xFFFFu) << 16);   // atomic_barrier=0 | dim0[15:0]
  g1[2] = (int)(((tensor_d0 >> 16) & 0xFFFFu) | ((tensor_d1 & 0xFFFFu) << 16));
  g1[3] = (int)(((tensor_d1 >> 16) & 0xFFFFu) | ((tile_d0 & 0xFFFFu) << 16));
  g1[4] = (int)(tile_d1 & 0xFFFFu);             // tile_dim1 | tile_dim2=0 (2D)
  g1[5] = (int)row_stride;                      // tensor_dim0_stride[31:0]
  g1[6] = 0;                                    // stride[47:32] | dim1_stride lo
  g1[7] = 0;
  v4i g2 = {};                                  // group 2 unused (2D tile)
  v4i g3 = {};                                  // group 3 unused (2D tile)
  v8i g4 = {};                                  // extra operand (clang-23 form)
  __builtin_amdgcn_tensor_load_to_lds(g0, g1, g2, g3, g4, 0);
}

__global__ void RB_f32_to_f16(const float* __restrict__ in, _Float16* __restrict__ out, int n) {
  int i = blockIdx.x * blockDim.x + threadIdx.x;
  if (i < n) out[i] = (_Float16)in[i];
}

// ---------------- QKV projection: [8192,1024] x [1024,3072] ------------------
// 256 threads = 8 waves; block tile 128(M) x 128(N); wave tile 32(M) x 64(N).
// Double-buffered: A tile via TDM (TENSORcnt), W tile via load-early/store-late.
__global__ __launch_bounds__(256)
void RB_qkv_gemm(const _Float16* __restrict__ X, const _Float16* __restrict__ W,
                 const float* __restrict__ bias,
                 _Float16* __restrict__ Qo, _Float16* __restrict__ Ko,
                 _Float16* __restrict__ Vo) {
  __shared__ _Float16 As[2][128 * LDS_STRIDE];
  __shared__ _Float16 Bs[2][128 * LDS_STRIDE];   // stored transposed: [n][k]

  const int tid = threadIdx.x;
  const int lane = tid & 31, w = tid >> 5;
  const int lane16 = lane & 15, hf = lane >> 4;
  const int wm = w & 3, wn = w >> 2;            // 4 waves along M, 2 along N
  const int m0 = blockIdx.y * 128, n0 = blockIdx.x * 128;
  const int kk = tid >> 3, c0b = (tid & 7) * 16;  // W staging coords

  v8f acc[2][4] = {};

  // ---- prologue: stage tile 0 into buffer 0 ----
  if (w == 0)
    tdm_load_2d(X + (size_t)m0 * D_MODEL, (uint32_t)(uintptr_t)&As[0][0],
                D_MODEL, BATCH * SEQ, 32, 128, D_MODEL, 1);
  {
    const _Float16* gp = W + (size_t)kk * NQKV + n0 + c0b;
    union { uint4 u; _Float16 e[8]; } a, b;
    a.u = *reinterpret_cast<const uint4*>(gp);
    b.u = *reinterpret_cast<const uint4*>(gp + 8);
#pragma unroll
    for (int j = 0; j < 8; ++j) Bs[0][(c0b + j) * LDS_STRIDE + kk] = a.e[j];
#pragma unroll
    for (int j = 0; j < 8; ++j) Bs[0][(c0b + 8 + j) * LDS_STRIDE + kk] = b.e[j];
  }
  if (w == 0) __builtin_amdgcn_s_wait_tensorcnt(0);
  __syncthreads();

#pragma unroll 1
  for (int i = 0; i < D_MODEL / 32; ++i) {
    const int cur = i & 1, nxt = cur ^ 1;
    const bool haveNext = (i + 1) < (D_MODEL / 32);
    union { uint4 u; _Float16 e[8]; } wa = {}, wb = {};
    if (haveNext) {
      const int k0n = (i + 1) * 32;
      const _Float16* gp = W + (size_t)(k0n + kk) * NQKV + n0 + c0b;
      wa.u = *reinterpret_cast<const uint4*>(gp);      // issue early (LOADcnt)
      wb.u = *reinterpret_cast<const uint4*>(gp + 8);
      if (w == 0)                                       // overlap TDM w/ compute
        tdm_load_2d(X + (size_t)m0 * D_MODEL + k0n, (uint32_t)(uintptr_t)&As[nxt][0],
                    D_MODEL, BATCH * SEQ, 32, 128, D_MODEL, 1);
    }

    // ---- compute on current buffer ----
    const _Float16* Ab = &As[cur][0];
    const _Float16* Bb = &Bs[cur][0];
    v16h afrag[2];
#pragma unroll
    for (int mt = 0; mt < 2; ++mt)
      afrag[mt] = load_ab_frag(&Ab[(wm * 32 + mt * 16 + lane16) * LDS_STRIDE + hf * 8]);
#pragma unroll
    for (int nt = 0; nt < 4; ++nt) {
      v16h bfrag = load_ab_frag(&Bb[(wn * 64 + nt * 16 + lane16) * LDS_STRIDE + hf * 8]);
#pragma unroll
      for (int mt = 0; mt < 2; ++mt)
        acc[mt][nt] = __builtin_amdgcn_wmma_f32_16x16x32_f16(
            false, afrag[mt], false, bfrag, (short)0, acc[mt][nt], false, false);
    }

    if (haveNext) {
      // transpose-store W tile into the other buffer (no conflict w/ compute)
#pragma unroll
      for (int j = 0; j < 8; ++j) Bs[nxt][(c0b + j) * LDS_STRIDE + kk] = wa.e[j];
#pragma unroll
      for (int j = 0; j < 8; ++j) Bs[nxt][(c0b + 8 + j) * LDS_STRIDE + kk] = wb.e[j];
      if (w == 0) __builtin_amdgcn_s_wait_tensorcnt(0);
    }
    __syncthreads();
  }

  // epilogue: + bias, scatter into q/k/v [B,H,T,64]; q pre-scaled by 1/sqrt(d)
  const float scale = 0.125f;
#pragma unroll
  for (int nt = 0; nt < 4; ++nt) {
    const int nG = n0 + wn * 64 + nt * 16 + lane16;
    const float bv = bias[nG];
    const int which = nG >> 10;
    const int col = nG & 1023;
    const int hd = col >> 6, dd = col & 63;
    _Float16* dst = (which == 0) ? Qo : ((which == 1) ? Ko : Vo);
    const float mul = (which == 0) ? scale : 1.0f;
#pragma unroll
    for (int mt = 0; mt < 2; ++mt) {
#pragma unroll
      for (int r = 0; r < 8; ++r) {
        const int mG = m0 + wm * 32 + mt * 16 + hf * 8 + r;
        const int bb = mG >> 11, tt = mG & (SEQ - 1);
        const float v = (acc[mt][nt][r] + bv) * mul;
        dst[((size_t)((bb * N_HEADS + hd) * SEQ + tt)) * HEAD_DIM + dd] = (_Float16)v;
      }
    }
  }
}

// ---------------- Flash attention with recency bias --------------------------
// 128 threads = 4 waves; block handles (b, h, 64 query rows); wave = 16 rows.
__global__ __launch_bounds__(128)
void RB_attn(const _Float16* __restrict__ Q, const _Float16* __restrict__ K,
             const _Float16* __restrict__ V, const float* __restrict__ rbias,
             const unsigned char* __restrict__ mask, _Float16* __restrict__ ctx) {
  __shared__ _Float16 Vt[HEAD_DIM * LDS_STRIDE];       // V^T tile: [d][key]
  __shared__ _Float16 Pbuf[4 * 16 * LDS_STRIDE];       // per-wave P transpose

  const int tid = threadIdx.x;
  const int lane = tid & 31, w = tid >> 5;
  const int lane16 = lane & 15, hf = lane >> 4;
  const int qb = blockIdx.x, hd = blockIdx.y, bb = blockIdx.z;
  const size_t headoff = (size_t)(bb * N_HEADS + hd) * SEQ * HEAD_DIM;
  const _Float16* qh = Q + headoff;
  const _Float16* kh = K + headoff;
  const _Float16* vh = V + headoff;
  const int q0 = qb * 64 + w * 16;

  // Q fragments (16 rows x 64 K, two 16x32 A fragments), kept in registers
  v16h aq[2];
#pragma unroll
  for (int kd = 0; kd < 2; ++kd)
    aq[kd] = load_ab_frag(qh + (size_t)(q0 + lane16) * HEAD_DIM + kd * 32 + hf * 8);

  v8f o[4] = {};
  float m_i[8], l_i[8];
#pragma unroll
  for (int r = 0; r < 8; ++r) { m_i[r] = -3.0e38f; l_i[r] = 0.0f; }

  _Float16* pb = &Pbuf[w * 16 * LDS_STRIDE];

#pragma unroll 1
  for (int kb = 0; kb < SEQ; kb += 32) {
    { // stage V block 32x64 transposed -> Vt[d][key]
      const int kl = tid >> 2, c0 = (tid & 3) * 16;
      const _Float16* gp = vh + (size_t)(kb + kl) * HEAD_DIM + c0;
      union { uint4 u; _Float16 e[8]; } a, b;
      a.u = *reinterpret_cast<const uint4*>(gp);
      b.u = *reinterpret_cast<const uint4*>(gp + 8);
#pragma unroll
      for (int j = 0; j < 8; ++j) Vt[(c0 + j) * LDS_STRIDE + kl] = a.e[j];
#pragma unroll
      for (int j = 0; j < 8; ++j) Vt[(c0 + 8 + j) * LDS_STRIDE + kl] = b.e[j];
    }
    __syncthreads();

    // S = (scaled Q) K^T : K row-major [T,64] is directly the B^T fragment src
    v8f s[2] = {};
#pragma unroll
    for (int st = 0; st < 2; ++st)
#pragma unroll
      for (int kd = 0; kd < 2; ++kd) {
        v16h bk = load_ab_frag(kh + (size_t)(kb + st * 16 + lane16) * HEAD_DIM + kd * 32 + hf * 8);
        s[st] = __builtin_amdgcn_wmma_f32_16x16x32_f16(
            false, aq[kd], false, bk, (short)0, s[st], false, false);
      }

    // + recency bias, key mask
    float sv[2][8];
#pragma unroll
    for (int st = 0; st < 2; ++st) {
      const int kg = kb + st * 16 + lane16;
      const bool keep = mask[bb * SEQ + kg] != 0;
#pragma unroll
      for (int r = 0; r < 8; ++r) {
        const int qg = q0 + hf * 8 + r;
        float vv = s[st][r] + rbias[(size_t)qg * SEQ + kg];
        sv[st][r] = keep ? vv : -3.0e38f;
      }
    }

    // online softmax (row lives across one 16-lane half -> xor masks 1..8)
    float ps[2][8];
#pragma unroll
    for (int r = 0; r < 8; ++r) {
      float mb = fmaxf(sv[0][r], sv[1][r]);
#pragma unroll
      for (int d = 1; d < 16; d <<= 1) mb = fmaxf(mb, __shfl_xor(mb, d, 32));
      const float mn = fmaxf(m_i[r], mb);
      const float alpha = __expf(m_i[r] - mn);
      const float p0 = __expf(sv[0][r] - mn);
      const float p1 = __expf(sv[1][r] - mn);
      ps[0][r] = p0; ps[1][r] = p1;
      float rs = p0 + p1;
#pragma unroll
      for (int d = 1; d < 16; d <<= 1) rs += __shfl_xor(rs, d, 32);
      l_i[r] = l_i[r] * alpha + rs;
      m_i[r] = mn;
#pragma unroll
      for (int nt = 0; nt < 4; ++nt) o[nt][r] *= alpha;
    }

    // C-layout P -> A-layout via wave-private LDS
#pragma unroll
    for (int st = 0; st < 2; ++st)
#pragma unroll
      for (int r = 0; r < 8; ++r)
        pb[(hf * 8 + r) * LDS_STRIDE + st * 16 + lane16] = (_Float16)ps[st][r];
    v16h ap = load_ab_frag(&pb[lane16 * LDS_STRIDE + hf * 8]);

    // O += P V
#pragma unroll
    for (int nt = 0; nt < 4; ++nt) {
      v16h bv = load_ab_frag(&Vt[(nt * 16 + lane16) * LDS_STRIDE + hf * 8]);
      o[nt] = __builtin_amdgcn_wmma_f32_16x16x32_f16(
          false, ap, false, bv, (short)0, o[nt], false, false);
    }
    __syncthreads();
  }

  // normalize and store ctx as [B,T,D] f16 (feeds out-proj GEMM)
#pragma unroll
  for (int r = 0; r < 8; ++r) {
    const float inv = (l_i[r] > 0.0f) ? (1.0f / l_i[r]) : 0.0f;
    const int tt = q0 + hf * 8 + r;
#pragma unroll
    for (int nt = 0; nt < 4; ++nt) {
      const int col = hd * HEAD_DIM + nt * 16 + lane16;
      ctx[((size_t)(bb * SEQ + tt)) * D_MODEL + col] = (_Float16)(o[nt][r] * inv);
    }
  }
}

// ---------------- Output projection: [8192,1024] x [1024,1024] ---------------
__global__ __launch_bounds__(256)
void RB_out_gemm(const _Float16* __restrict__ X, const _Float16* __restrict__ W,
                 const float* __restrict__ bias, float* __restrict__ out) {
  __shared__ _Float16 As[2][128 * LDS_STRIDE];
  __shared__ _Float16 Bs[2][128 * LDS_STRIDE];

  const int tid = threadIdx.x;
  const int lane = tid & 31, w = tid >> 5;
  const int lane16 = lane & 15, hf = lane >> 4;
  const int wm = w & 3, wn = w >> 2;
  const int m0 = blockIdx.y * 128, n0 = blockIdx.x * 128;
  const int kk = tid >> 3, c0b = (tid & 7) * 16;

  v8f acc[2][4] = {};

  if (w == 0)
    tdm_load_2d(X + (size_t)m0 * D_MODEL, (uint32_t)(uintptr_t)&As[0][0],
                D_MODEL, BATCH * SEQ, 32, 128, D_MODEL, 1);
  {
    const _Float16* gp = W + (size_t)kk * D_MODEL + n0 + c0b;
    union { uint4 u; _Float16 e[8]; } a, b;
    a.u = *reinterpret_cast<const uint4*>(gp);
    b.u = *reinterpret_cast<const uint4*>(gp + 8);
#pragma unroll
    for (int j = 0; j < 8; ++j) Bs[0][(c0b + j) * LDS_STRIDE + kk] = a.e[j];
#pragma unroll
    for (int j = 0; j < 8; ++j) Bs[0][(c0b + 8 + j) * LDS_STRIDE + kk] = b.e[j];
  }
  if (w == 0) __builtin_amdgcn_s_wait_tensorcnt(0);
  __syncthreads();

#pragma unroll 1
  for (int i = 0; i < D_MODEL / 32; ++i) {
    const int cur = i & 1, nxt = cur ^ 1;
    const bool haveNext = (i + 1) < (D_MODEL / 32);
    union { uint4 u; _Float16 e[8]; } wa = {}, wb = {};
    if (haveNext) {
      const int k0n = (i + 1) * 32;
      const _Float16* gp = W + (size_t)(k0n + kk) * D_MODEL + n0 + c0b;
      wa.u = *reinterpret_cast<const uint4*>(gp);
      wb.u = *reinterpret_cast<const uint4*>(gp + 8);
      if (w == 0)
        tdm_load_2d(X + (size_t)m0 * D_MODEL + k0n, (uint32_t)(uintptr_t)&As[nxt][0],
                    D_MODEL, BATCH * SEQ, 32, 128, D_MODEL, 1);
    }

    const _Float16* Ab = &As[cur][0];
    const _Float16* Bb = &Bs[cur][0];
    v16h afrag[2];
#pragma unroll
    for (int mt = 0; mt < 2; ++mt)
      afrag[mt] = load_ab_frag(&Ab[(wm * 32 + mt * 16 + lane16) * LDS_STRIDE + hf * 8]);
#pragma unroll
    for (int nt = 0; nt < 4; ++nt) {
      v16h bfrag = load_ab_frag(&Bb[(wn * 64 + nt * 16 + lane16) * LDS_STRIDE + hf * 8]);
#pragma unroll
      for (int mt = 0; mt < 2; ++mt)
        acc[mt][nt] = __builtin_amdgcn_wmma_f32_16x16x32_f16(
            false, afrag[mt], false, bfrag, (short)0, acc[mt][nt], false, false);
    }

    if (haveNext) {
#pragma unroll
      for (int j = 0; j < 8; ++j) Bs[nxt][(c0b + j) * LDS_STRIDE + kk] = wa.e[j];
#pragma unroll
      for (int j = 0; j < 8; ++j) Bs[nxt][(c0b + 8 + j) * LDS_STRIDE + kk] = wb.e[j];
      if (w == 0) __builtin_amdgcn_s_wait_tensorcnt(0);
    }
    __syncthreads();
  }

#pragma unroll
  for (int nt = 0; nt < 4; ++nt) {
    const int nG = n0 + wn * 64 + nt * 16 + lane16;
    const float bv = bias[nG];
#pragma unroll
    for (int mt = 0; mt < 2; ++mt) {
#pragma unroll
      for (int r = 0; r < 8; ++r) {
        const int mG = m0 + wm * 32 + mt * 16 + hf * 8 + r;
        out[(size_t)mG * D_MODEL + nG] = acc[mt][nt][r] + bv;
      }
    }
  }
}

extern "C" void kernel_launch(void* const* d_in, const int* in_sizes, int n_in,
                              void* d_out, int out_size, void* d_ws, size_t ws_size,
                              hipStream_t stream) {
  (void)in_sizes; (void)n_in; (void)out_size; (void)ws_size;
  const float*         x     = (const float*)d_in[0];
  const unsigned char* mask  = (const unsigned char*)d_in[1];
  const float*         Wqkv  = (const float*)d_in[2];
  const float*         bqkv  = (const float*)d_in[3];
  const float*         Wout  = (const float*)d_in[4];
  const float*         bout  = (const float*)d_in[5];
  const float*         rbias = (const float*)d_in[6];
  float*               out   = (float*)d_out;

  char* ws = (char*)d_ws;
  const size_t szX    = (size_t)BATCH * SEQ * D_MODEL * 2;            // 16 MB
  const size_t szWq   = (size_t)D_MODEL * NQKV * 2;                   //  6 MB
  const size_t szWo   = (size_t)D_MODEL * D_MODEL * 2;                //  2 MB
  const size_t szHead = (size_t)BATCH * N_HEADS * SEQ * HEAD_DIM * 2; // 16 MB
  _Float16* xh    = (_Float16*)(ws);
  _Float16* wqkvh = (_Float16*)(ws + szX);
  _Float16* wouth = (_Float16*)(ws + szX + szWq);
  _Float16* qh    = (_Float16*)(ws + szX + szWq + szWo);
  _Float16* kh    = (_Float16*)(ws + szX + szWq + szWo + szHead);
  _Float16* vh    = (_Float16*)(ws + szX + szWq + szWo + 2 * szHead);
  _Float16* ctxh  = (_Float16*)(ws + szX + szWq + szWo + 3 * szHead);

  const int nX  = BATCH * SEQ * D_MODEL;     // 8388608
  const int nWq = D_MODEL * NQKV;            // 3145728
  const int nWo = D_MODEL * D_MODEL;         // 1048576
  RB_f32_to_f16<<<(nX + 255) / 256, 256, 0, stream>>>(x, xh, nX);
  RB_f32_to_f16<<<(nWq + 255) / 256, 256, 0, stream>>>(Wqkv, wqkvh, nWq);
  RB_f32_to_f16<<<(nWo + 255) / 256, 256, 0, stream>>>(Wout, wouth, nWo);

  RB_qkv_gemm<<<dim3(NQKV / 128, (BATCH * SEQ) / 128), 256, 0, stream>>>(
      xh, wqkvh, bqkv, qh, kh, vh);

  RB_attn<<<dim3(SEQ / 64, N_HEADS, BATCH), 128, 0, stream>>>(
      qh, kh, vh, rbias, mask, ctxh);

  RB_out_gemm<<<dim3(D_MODEL / 128, (BATCH * SEQ) / 128), 256, 0, stream>>>(
      ctxh, wouth, bout, out);
}
